// MultiHeadLatentAttention_45638322487940
// MI455X (gfx1250) — matrix operand order repaired and checked by reading, not compiled, not run on hardware
//
#include <hip/hip_runtime.h>
#include <hip/hip_bf16.h>

typedef __attribute__((ext_vector_type(16))) _Float16 v16h;
typedef __attribute__((ext_vector_type(8)))  _Float16 v8h;
typedef __attribute__((ext_vector_type(8)))  float    v8f;

#define BB  2
#define TT  2048
#define CC  1024
#define HH  16
#define HD  64
#define KVD 16
#define MM  (BB * TT)

// ---------------------------------------------------------------------------
// Helpers: 16B vector loads, zero fill, v8h+v8h -> v16h fragment concat.
// ---------------------------------------------------------------------------
__device__ __forceinline__ v8h zero8h() {
  v8h z;
  #pragma unroll
  for (int i = 0; i < 8; ++i) z[i] = (_Float16)0.0f;
  return z;
}
__device__ __forceinline__ v16h zero16h() {
  v16h z;
  #pragma unroll
  for (int i = 0; i < 16; ++i) z[i] = (_Float16)0.0f;
  return z;
}
__device__ __forceinline__ v8h ld8(const _Float16* p) { return *(const v8h*)p; }
__device__ __forceinline__ v16h cat16(v8h lo, v8h hi) {
  return __builtin_shufflevector(lo, hi, 0, 1, 2, 3, 4, 5, 6, 7,
                                 8, 9, 10, 11, 12, 13, 14, 15);
}

// ---------------------------------------------------------------------------
// f32 -> f16 elementwise conversion
// ---------------------------------------------------------------------------
__global__ __launch_bounds__(256)
void cvt_f32_to_f16(const float* __restrict__ src, _Float16* __restrict__ dst, int n) {
  int i = blockIdx.x * blockDim.x + threadIdx.x;
  if (i < n) dst[i] = (_Float16)src[i];
}

// f32 [K,N] -> f16 transposed [N,K]; 2D indexing (no integer div/mod).
__global__ __launch_bounds__(256)
void cvt_transpose_f16(const float* __restrict__ src, _Float16* __restrict__ dst,
                       int K, int N) {
  int n = blockIdx.x * 16 + (int)threadIdx.x % 16;
  int k = blockIdx.y * 16 + (int)threadIdx.x / 16;
  if (k < K && n < N)
    dst[(size_t)n * K + k] = (_Float16)src[(size_t)k * N + n];
}

// ---------------------------------------------------------------------------
// Register-resident WMMA GEMM: out[M,N] = A[M,K] @ WT[N,K]^T + bias[N].
// K and OUTMODE are compile-time so all bounds guards fold away:
//   K % 32 == 0 -> unguarded immediate-offset global_load_b128 loop
//   K == 16     -> single step, OOB runs are compile-time zeros / cndmask
// Block = 256 threads = 8 waves; tile 64x64; wave owns 16 rows x 32 cols.
// OUTMODE: 0 = f32 [M,N], 1 = f16 [M,N], 2 = f16 [b][h][d][t] (V-transposed).
// ---------------------------------------------------------------------------
template <int K, int OUTMODE>
__global__ __launch_bounds__(256)
void gemm_wmma_tn(const _Float16* __restrict__ A, const _Float16* __restrict__ WT,
                  const float* __restrict__ bias, void* __restrict__ outp,
                  int M, int N) {
  const int lane = threadIdx.x & 31;
  const int wave = threadIdx.x >> 5;          // 0..7
  const int lHi  = lane >> 4;                 // 0/1
  const int lLo  = lane & 15;
  const int wy   = wave >> 1;                 // 0..3 -> 16-row slice
  const int wx   = wave & 1;                  // 0..1 -> 32-col slice
  const int M0   = blockIdx.y << 6;
  const int N0   = blockIdx.x << 6;

  const int row = M0 + (wy << 4) + lLo;       // this lane's A-frag row
  const int n0  = N0 + (wx << 5) + lLo;       // B-frag col, subtile 0
  const int n1  = n0 + 16;                    // B-frag col, subtile 1
  // Clamp OOB rows to a valid address; their results are never stored.
  const _Float16* arow  = A  + (size_t)(row < M ? row : M - 1) * K + (lHi << 3);
  const _Float16* wrow0 = WT + (size_t)(n0 < N ? n0 : N - 1) * K + (lHi << 3);
  const _Float16* wrow1 = WT + (size_t)(n1 < N ? n1 : N - 1) * K + (lHi << 3);

  v8f acc0, acc1;
  #pragma unroll
  for (int j = 0; j < 8; ++j) { acc0[j] = 0.0f; acc1[j] = 0.0f; }

  if constexpr (K == 16) {
    // Single K-step. A-frag: lower run (K = lHi*8 + e, always < 16) loaded,
    // upper run (K >= 16) is a compile-time zero. B-frag: same run structure
    // (K = lHi*8 + e loaded, upper run zero).
    const _Float16* wb0 = WT + (size_t)(n0 < N ? n0 : N - 1) * K;
    const _Float16* wb1 = WT + (size_t)(n1 < N ? n1 : N - 1) * K;
    v16h af  = cat16(ld8(arow), zero8h());
    v16h bf0 = cat16(ld8(wb0 + (lHi << 3)), zero8h());
    v16h bf1 = cat16(ld8(wb1 + (lHi << 3)), zero8h());
    acc0 = __builtin_amdgcn_wmma_f32_16x16x32_f16(false, af, false, bf0,
                                                  (short)0, acc0, false, false);
    acc1 = __builtin_amdgcn_wmma_f32_16x16x32_f16(false, af, false, bf1,
                                                  (short)0, acc1, false, false);
  } else {
    static_assert(K % 32 == 0, "K must be a multiple of 32");
    #pragma unroll 4
    for (int kk = 0; kk < K; kk += 32) {
      // A: M=lLo, K = {lHi*8+e, 16+lHi*8+e}; B: N=col, K = {lHi*8+e, 16+lHi*8+e}.
      v16h af  = cat16(ld8(arow + kk),      ld8(arow + kk + 16));
      v16h bf0 = cat16(ld8(wrow0 + kk),     ld8(wrow0 + kk + 16));
      v16h bf1 = cat16(ld8(wrow1 + kk),     ld8(wrow1 + kk + 16));
      if (kk + 32 < K)                      // global_prefetch_b8 next K-step
        __builtin_prefetch(arow + kk + 32, 0, 1);
      acc0 = __builtin_amdgcn_wmma_f32_16x16x32_f16(false, af, false, bf0,
                                                    (short)0, acc0, false, false);
      acc1 = __builtin_amdgcn_wmma_f32_16x16x32_f16(false, af, false, bf1,
                                                    (short)0, acc1, false, false);
    }
  }

  #pragma unroll
  for (int s = 0; s < 2; ++s) {
    const v8f acc = s ? acc1 : acc0;
    #pragma unroll
    for (int j = 0; j < 8; ++j) {
      const int m = M0 + (wy << 4) + (lHi << 3) + j;   // D: M=(lane/16)*8+j
      const int n = N0 + (wx << 5) + (s << 4) + lLo;   // D: N=lane%16
      if (m < M && n < N) {
        const float v = acc[j] + bias[n];
        if constexpr (OUTMODE == 0) {
          ((float*)outp)[(size_t)m * N + n] = v;
        } else if constexpr (OUTMODE == 1) {
          ((_Float16*)outp)[(size_t)m * N + n] = (_Float16)v;
        } else {  // V-transposed layout [b][h][d][t] for attention B-frags
          const int bI = m >> 11, tI = m & (TT - 1);
          const int hI = n >> 6,  dI = n & (HD - 1);
          ((_Float16*)outp)[(((size_t)bI * HH + hI) * HD + dI) * TT + tI] =
              (_Float16)v;
        }
      }
    }
  }
}

// ---------------------------------------------------------------------------
// Causal flash attention, one wave per 16-query tile per (batch, head).
// K^T and V fragments load straight from global as b128 vectors (K row-major
// along hd; V pre-transposed to [b][h][d][t]). LDS only for softmax spill,
// P repack, and per-row rescale broadcast.
// ---------------------------------------------------------------------------
__global__ __launch_bounds__(32)
void mla_flash_attn(const _Float16* __restrict__ qh, const _Float16* __restrict__ kh,
                    const _Float16* __restrict__ vT, _Float16* __restrict__ yh) {
  const int qt   = blockIdx.x;
  const int h    = blockIdx.y;
  const int b    = blockIdx.z;
  const int lane = threadIdx.x;
  const int lHi  = lane >> 4;
  const int lLo  = lane & 15;

  __shared__ float Ssh[16][33];                    // 16 q x 32 key scores
  __shared__ __align__(16) _Float16 Psh[16][40];   // exp probs (16B rows)
  __shared__ float rowFix[16];

  // Q fragments (hd chunks 0..31, 32..63), two aligned b128 loads each.
  const _Float16* qrow = qh + (size_t)(b * TT + qt * 16 + lLo) * CC + h * HD;
  v16h qf[2];
  #pragma unroll
  for (int c = 0; c < 2; ++c)
    qf[c] = cat16(ld8(qrow + c * 32 + (lHi << 3)),
                  ld8(qrow + c * 32 + 16 + (lHi << 3)));

  v8f o[4];
  #pragma unroll
  for (int g = 0; g < 4; ++g)
    #pragma unroll
    for (int j = 0; j < 8; ++j) o[g][j] = 0.0f;

  const _Float16* kbase = kh + (size_t)b * TT * CC + (size_t)h * HD;
  const _Float16* vbase = vT + ((size_t)b * HH + h) * HD * TT;

  const float scale = 0.125f;                 // 1/sqrt(64)
  float rMax = -__builtin_inff();             // per-row stats on lanes 0..15
  float rSum = 0.0f;

  const int qEnd   = qt * 16 + 15;
  const int nTiles = qEnd / 32 + 1;           // causal tile range

  for (int jt = 0; jt < nTiles; ++jt) {
    const int key0 = jt * 32;

    // S tile: two 16-key subtiles x two hd-32 chunks; B = K^T directly
    // from global (N=key=sub*16+lLo, K=hd -> contiguous along hd).
    #pragma unroll
    for (int sub = 0; sub < 2; ++sub) {
      const _Float16* krow = kbase + (size_t)(key0 + sub * 16 + lLo) * CC;
      v8f s;
      #pragma unroll
      for (int j = 0; j < 8; ++j) s[j] = 0.0f;
      #pragma unroll
      for (int c = 0; c < 2; ++c) {
        v16h bk = cat16(ld8(krow + c * 32 + (lHi << 3)),
                        ld8(krow + c * 32 + (lHi << 3) + 16));
        s = __builtin_amdgcn_wmma_f32_16x16x32_f16(false, qf[c], false, bk,
                                                   (short)0, s, false, false);
      }
      #pragma unroll
      for (int j = 0; j < 8; ++j)
        Ssh[(lHi << 3) + j][sub * 16 + lLo] = s[j] * scale;
    }
    __syncthreads();

    // Online softmax row update (row r owned by lane r, r < 16).
    if (lane < 16) {
      const int qpos = qt * 16 + lane;
      float sv[32];
      float tmax = -__builtin_inff();
      #pragma unroll
      for (int c = 0; c < 32; ++c) {
        float s = ((key0 + c) <= qpos) ? Ssh[lane][c] : -__builtin_inff();
        sv[c] = s;
        tmax = fmaxf(tmax, s);
      }
      const float nMax = fmaxf(rMax, tmax);
      const float fix  = __expf(rMax - nMax);   // 0 on first tile
      float tsum = 0.0f;
      #pragma unroll
      for (int c = 0; c < 32; ++c) {
        float p = ((key0 + c) <= qpos) ? __expf(sv[c] - nMax) : 0.0f;
        tsum += p;
        Psh[lane][c] = (_Float16)p;
      }
      rSum = rSum * fix + tsum;
      rMax = nMax;
      rowFix[lane] = fix;
    }
    __syncthreads();

    // P A-fragment: two aligned ds b128 loads (same run structure as A).
    v16h pf = cat16(ld8(&Psh[lLo][(lHi << 3)]),
                    ld8(&Psh[lLo][16 + (lHi << 3)]));
    // O rescale + O += P V; B = V from transposed global layout
    // (K=key={lHi*8+e, 16+lHi*8+e} along t, N=hd=g*16+lLo).
    #pragma unroll
    for (int g = 0; g < 4; ++g) {
      const _Float16* vrow = vbase + (size_t)((g << 4) + lLo) * TT + key0;
      v16h bv = cat16(ld8(vrow + (lHi << 3)),
                      ld8(vrow + (lHi << 3) + 16));
      #pragma unroll
      for (int j = 0; j < 8; ++j)
        o[g][j] *= rowFix[(lHi << 3) + j];
      o[g] = __builtin_amdgcn_wmma_f32_16x16x32_f16(false, pf, false, bv,
                                                    (short)0, o[g], false, false);
    }
    __syncthreads();
  }

  // Normalize and store f16 output rows for this head.
  if (lane < 16) rowFix[lane] = 1.0f / rSum;
  __syncthreads();
  #pragma unroll
  for (int g = 0; g < 4; ++g)
    #pragma unroll
    for (int j = 0; j < 8; ++j) {
      const int m = (lHi << 3) + j;
      const size_t idx = (size_t)(b * TT + qt * 16 + m) * CC +
                         (size_t)h * HD + g * 16 + lLo;
      yh[idx] = (_Float16)(o[g][j] * rowFix[m]);
    }
}

// ---------------------------------------------------------------------------
// Launch: f16 conversions (weights transposed), 4 GEMMs, attention, out GEMM.
// ---------------------------------------------------------------------------
extern "C" void kernel_launch(void* const* d_in, const int* in_sizes, int n_in,
                              void* d_out, int out_size, void* d_ws, size_t ws_size,
                              hipStream_t stream) {
  (void)in_sizes; (void)n_in; (void)out_size; (void)ws_size;
  const float* x  = (const float*)d_in[0];
  const float* Wc = (const float*)d_in[1];
  const float* bc = (const float*)d_in[2];
  const float* Wk = (const float*)d_in[3];
  const float* bk = (const float*)d_in[4];
  const float* Wv = (const float*)d_in[5];
  const float* bv = (const float*)d_in[6];
  const float* Wq = (const float*)d_in[7];
  const float* bq = (const float*)d_in[8];
  const float* Wo = (const float*)d_in[9];
  const float* bo = (const float*)d_in[10];
  // d_in[11] = mask: tril(ones) by construction; causality computed analytically.

  char* ws = (char*)d_ws;
  size_t off = 0;
  auto take = [&](size_t nElem) -> _Float16* {
    _Float16* p = (_Float16*)(ws + off);
    off += ((nElem * sizeof(_Float16)) + 255) & ~(size_t)255;
    return p;
  };
  _Float16* xh  = take((size_t)MM * CC);
  _Float16* WqT = take((size_t)CC * CC);    // [N=C][K=C]
  _Float16* WoT = take((size_t)CC * CC);
  _Float16* WcT = take((size_t)KVD * CC);   // [N=16][K=C]
  _Float16* WkT = take((size_t)CC * KVD);   // [N=C][K=16]
  _Float16* WvT = take((size_t)CC * KVD);
  _Float16* kvh = take((size_t)MM * KVD);
  _Float16* qh  = take((size_t)MM * CC);
  _Float16* khb = take((size_t)MM * CC);
  _Float16* vTb = take((size_t)MM * CC);    // [b][h][d][t]
  _Float16* yh  = take((size_t)MM * CC);

  cvt_f32_to_f16<<<(MM * CC + 255) / 256, 256, 0, stream>>>(x, xh, MM * CC);
  auto cvtT = [&](const float* s, _Float16* d, int K, int N) {
    cvt_transpose_f16<<<dim3((N + 15) / 16, (K + 15) / 16), 256, 0, stream>>>(s, d, K, N);
  };
  cvtT(Wq, WqT, CC, CC);
  cvtT(Wo, WoT, CC, CC);
  cvtT(Wc, WcT, CC, KVD);
  cvtT(Wk, WkT, KVD, CC);
  cvtT(Wv, WvT, KVD, CC);

  const dim3 blk(256);
  // kv_compressed = x @ Wc + bc          [M x 16] f16
  gemm_wmma_tn<CC, 1><<<dim3(1, MM / 64), blk, 0, stream>>>(xh, WcT, bc, kvh, MM, KVD);
  // q = x @ Wq + bq                      [M x C] f16
  gemm_wmma_tn<CC, 1><<<dim3(CC / 64, MM / 64), blk, 0, stream>>>(xh, WqT, bq, qh, MM, CC);
  // k = kv @ Wk + bk                     [M x C] f16 row-major
  gemm_wmma_tn<KVD, 1><<<dim3(CC / 64, MM / 64), blk, 0, stream>>>(kvh, WkT, bk, khb, MM, CC);
  // v = kv @ Wv + bv                     f16, stored transposed [b][h][d][t]
  gemm_wmma_tn<KVD, 2><<<dim3(CC / 64, MM / 64), blk, 0, stream>>>(kvh, WvT, bv, vTb, MM, CC);
  // y = softmax(q k^T / sqrt(hd), causal) v
  mla_flash_attn<<<dim3(TT / 16, HH, BB), dim3(32), 0, stream>>>(qh, khb, vTb, yh);
  // out = y @ Wo + bo                    [M x C] f32
  gemm_wmma_tn<CC, 0><<<dim3(CC / 64, MM / 64), blk, 0, stream>>>(yh, WoT, bo, d_out, MM, CC);
}
